// ScreenCorrespondence_40870908789100
// MI455X (gfx1250) — matrix operand order repaired and checked by reading, not compiled, not run on hardware
//
#include <hip/hip_runtime.h>

// ---------------- problem constants (fixed by the reference) ----------------
#define BSZ  4
#define GG   512
#define SS   1536        // 3*G
#define DM   256
#define HH   4
#define DH   64
#define DFF_ 1024
#define NTOK (BSZ*SS)    // 6144
#define NROW (BSZ*GG)    // 2048
#define VD_  25088
#define TD_  768
#define NL_  4
#define NB_  32          // num buckets

typedef __attribute__((ext_vector_type(16))) __bf16 v16bf;
typedef __attribute__((ext_vector_type(8)))  __bf16 v8bf;
typedef __attribute__((ext_vector_type(8)))  float  v8f;

// ---------------------------------------------------------------------------
// CDNA5 async global->LDS 16-byte copy (ASYNCcnt tracked) + waits
__device__ __forceinline__ void async_ld16(__bf16* lds, const __bf16* g) {
    asm volatile("global_load_async_to_lds_b128 %0, %1, off"
                 :: "v"((unsigned)(uintptr_t)lds), "v"(g) : "memory");
}
#define WAIT_ASYNC(n) asm volatile("s_wait_asynccnt " #n ::: "memory")

// ---------------------------------------------------------------------------
__device__ __forceinline__ int rp_bucket(float d) {
    float s = (d > 0.f) ? 1.f : ((d < 0.f) ? -1.f : 0.f);
    float n = fminf(fmaxf(fabsf(d), 1e-6f), 128.f);
    int li = (int)floorf(__log2f(n));
    li = min(max(li, 0), NB_ - 1);
    return li * (int)s + (NB_ - 1);
}

// ---------------------------------------------------------------------------
// f32 -> bf16 bulk convert
__global__ __launch_bounds__(256)
void cvt_bf16_kernel(const float* __restrict__ in, __bf16* __restrict__ out, int n)
{
    int i = blockIdx.x * 256 + threadIdx.x;
    if (i < n) out[i] = (__bf16)in[i];
}

// W (K x N, f32) -> Wt (N x K, bf16)   [coalesced writes]
__global__ __launch_bounds__(256)
void transpose_bf16_kernel(const float* __restrict__ W, __bf16* __restrict__ Wt,
                           int K, int N)
{
    int i = blockIdx.x * 256 + threadIdx.x;
    if (i < K * N) {
        int n = i / K, k = i % K;
        Wt[i] = (__bf16)W[(size_t)k * N + n];
    }
}

// ---------------------------------------------------------------------------
// prep: cc centers, no_text flag, text mask (one block per (b,g) row)
__global__ __launch_bounds__(256)
void prep_kernel(const float* __restrict__ coords, const float* __restrict__ texts,
                 const unsigned char* __restrict__ pm,
                 float* __restrict__ cc, float* __restrict__ tmask, float* __restrict__ ntf)
{
    __shared__ int anynz;
    int r = blockIdx.x, t = threadIdx.x;
    if (t == 0) anynz = 0;
    __syncthreads();
    const float* tp = texts + (size_t)r * TD_;
    float a0 = tp[t], a1 = tp[t + 256], a2 = tp[t + 512];
    if (a0 != 0.f || a1 != 0.f || a2 != 0.f) anynz = 1;
    __syncthreads();
    if (t == 0) {
        int notext = (anynz == 0);
        ntf[r] = notext ? 1.f : 0.f;
        tmask[r] = (pm[r] != 0 && !notext) ? 1.f : 0.f;
        cc[2*r+0] = (coords[(size_t)r*4+0] + coords[(size_t)r*4+2]) * 0.5f;
        cc[2*r+1] = (coords[(size_t)r*4+1] + coords[(size_t)r*4+3]) * 0.5f;
    }
}

__global__ __launch_bounds__(256)
void mask_kernel(const unsigned char* __restrict__ pm, const float* __restrict__ tmask,
                 float* __restrict__ maskbuf)
{
    int i = blockIdx.x * 256 + threadIdx.x;
    int b = i / SS, tk = i % SS, g = tk % GG;
    float mv = (tk < 2*GG) ? (pm[b*GG + g] ? 1.f : 0.f) : tmask[b*GG + g];
    maskbuf[i] = mv;
}

__global__ __launch_bounds__(256)
void gather_kernel(const int* __restrict__ class_idx, const float* __restrict__ class_emb,
                   float* __restrict__ x)
{
    int r = blockIdx.x, d = threadIdx.x;
    int b = r / GG, g = r % GG;
    int ci = class_idx[r];
    x[((size_t)b*SS + g)*DM + d] = class_emb[(size_t)ci*DM + d];
}

// ---------------------------------------------------------------------------
// bf16 WMMA GEMM, pre-transposed weights, async-to-LDS double-buffered.
//   A  : M x K bf16 (row-major)
//   Wt : N x K bf16 (weight pre-transposed -> B tiles are straight copies)
//   C  : f32 out (ld N) with optional accum/rowscale, or Cb : bf16 out
// Block = 256 threads (8 waves). Tile 64x64: wave (wr = wid>>1) rows,
// (wid&1)*32 cols, 2 accumulators. Every wave issues exactly 2 async b128
// per k-step -> uniform s_wait_asynccnt immediates.
__global__ __launch_bounds__(256)
void gemm_bf16t(const __bf16* __restrict__ A, const __bf16* __restrict__ Wt,
                const float* __restrict__ bias,
                float* __restrict__ C, __bf16* __restrict__ Cb,
                int M, int K, int N,
                int seg_in, int seg_out, int tok_off,
                const float* __restrict__ rowscale, int accum, int relu)
{
    __shared__ __bf16 As[2][64][40];   // [row][k] k-contiguous, 16B chunks
    __shared__ __bf16 Bs[2][64][40];   // [n][k]   k-contiguous

    const int t    = threadIdx.x;
    const int lane = t & 31;
    const int wid  = t >> 5;
    const int wr   = wid >> 1;          // 0..3 : 16-row block
    const int wcb  = (wid & 1) * 32;    // 0 / 32 : 32-col block
    const int lh   = lane >> 4;
    const int lm   = lane & 15;
    const int m0   = blockIdx.x * 64;
    const int n0   = blockIdx.y * 64;

    const int crow = t >> 2;            // 0..63
    const int ccol = (t & 3) * 8;       // 0,8,16,24 (halfs)

    const __bf16* agbase = A  + (size_t)(m0 + crow) * K + ccol;
    const __bf16* bgbase = Wt + (size_t)(n0 + crow) * K + ccol;

    v8f acc0 = {}, acc1 = {};

    const int nk = K >> 5;
    // prologue: tile 0 -> buffer 0
    async_ld16(&As[0][crow][ccol], agbase);
    async_ld16(&Bs[0][crow][ccol], bgbase);

    for (int kk = 0; kk < nk; ++kk) {
        const int cur = kk & 1;
        if (kk + 1 < nk) {
            const int k0n = (kk + 1) << 5;
            async_ld16(&As[cur ^ 1][crow][ccol], agbase + k0n);
            async_ld16(&Bs[cur ^ 1][crow][ccol], bgbase + k0n);
            WAIT_ASYNC(2);          // previous tile landed; prefetch in flight
        } else {
            WAIT_ASYNC(0);
        }
        __syncthreads();

        v8bf alo = *(const v8bf*)&As[cur][wr*16 + lm][lh*8];
        v8bf ahi = *(const v8bf*)&As[cur][wr*16 + lm][16 + lh*8];
        v16bf a;
        #pragma unroll
        for (int i = 0; i < 8; ++i) { a[i] = alo[i]; a[8+i] = ahi[i]; }

        #pragma unroll
        for (int j = 0; j < 2; ++j) {
            v8bf blo = *(const v8bf*)&Bs[cur][wcb + j*16 + lm][lh*8];
            v8bf bhi = *(const v8bf*)&Bs[cur][wcb + j*16 + lm][16 + lh*8];
            v16bf b;
            #pragma unroll
            for (int i = 0; i < 8; ++i) { b[i] = blo[i]; b[8+i] = bhi[i]; }
            if (j == 0)
                acc0 = __builtin_amdgcn_wmma_f32_16x16x32_bf16(false, a, false, b, (short)0, acc0, false, false);
            else
                acc1 = __builtin_amdgcn_wmma_f32_16x16x32_bf16(false, a, false, b, (short)0, acc1, false, false);
        }
        __syncthreads();
    }

    #pragma unroll
    for (int j = 0; j < 2; ++j) {
        const int col = n0 + wcb + j*16 + lm;
        const float bval = bias ? bias[col] : 0.f;
        const v8f* accp = j ? &acc1 : &acc0;
        #pragma unroll
        for (int r = 0; r < 8; ++r) {
            int m = m0 + wr*16 + r + 8*lh;
            int orow = (m / seg_in) * seg_out + (m % seg_in) + tok_off;
            float v = (*accp)[r] + bval;
            if (relu) v = fmaxf(v, 0.f);
            if (rowscale) v *= rowscale[m];
            if (Cb) {
                Cb[(size_t)orow * N + col] = (__bf16)v;
            } else {
                float* cp = C + (size_t)orow * N + col;
                if (accum) v += *cp;
                *cp = v;
            }
        }
    }
}

// ---------------------------------------------------------------------------
// LayerNorm over D=256 (one block per token), bf16 output for GEMM consumption
__global__ __launch_bounds__(256)
void layernorm_kernel(const float* __restrict__ x, const float* __restrict__ s,
                      const float* __restrict__ bsh, __bf16* __restrict__ out)
{
    __shared__ float red[256];
    int row = blockIdx.x, d = threadIdx.x;
    float v = x[(size_t)row * DM + d];
    red[d] = v; __syncthreads();
    #pragma unroll
    for (int off = 128; off > 0; off >>= 1) { if (d < off) red[d] += red[d + off]; __syncthreads(); }
    float mean = red[0] * (1.f / DM);
    __syncthreads();
    float c = v - mean;
    red[d] = c * c; __syncthreads();
    #pragma unroll
    for (int off = 128; off > 0; off >>= 1) { if (d < off) red[d] += red[d + off]; __syncthreads(); }
    float var = red[0] * (1.f / DM);
    out[(size_t)row * DM + d] = (__bf16)(c * rsqrtf(var + 1e-5f) * s[d] + bsh[d]);
}

// ---------------------------------------------------------------------------
// Flash-style attention, bf16 Q/K/V, K tiles staged via async-to-LDS.
__global__ __launch_bounds__(128)
void attn_kernel(const __bf16* __restrict__ Q, const __bf16* __restrict__ Kb,
                 const __bf16* __restrict__ Vb, __bf16* __restrict__ O,
                 const float* __restrict__ cc, const float* __restrict__ maskbuf,
                 const float* __restrict__ rpex, const float* __restrict__ rpey)
{
    __shared__ __bf16 Ks[32][72];        // K tile [kidx][dh]
    __shared__ __bf16 Vt[64][40];        // V^T   [dh][kidx]
    __shared__ __bf16 Pst[4][16][40];    // per-wave P staging
    __shared__ float  ccQ[64][2];
    __shared__ float  ccK[32][2];
    __shared__ float  mk[32];

    const int t = threadIdx.x;
    const int lane = t & 31, wid = t >> 5;
    const int lh = lane >> 4, lm = lane & 15;
    const int qb = blockIdx.x, h = blockIdx.y, b = blockIdx.z;
    const int q0 = qb * 64;

    if (t < 64) {
        int g = (q0 + t) % GG;
        ccQ[t][0] = cc[2*(b*GG + g) + 0];
        ccQ[t][1] = cc[2*(b*GG + g) + 1];
    }

    // Q fragments (A layout), loaded once
    const __bf16* qp = Q + (size_t)(b*SS + q0 + wid*16 + lm) * DM + h*DH;
    v16bf qa0, qa1;
    {
        v8bf q00 = *(const v8bf*)(qp +      lh*8);
        v8bf q01 = *(const v8bf*)(qp + 16 + lh*8);
        v8bf q10 = *(const v8bf*)(qp + 32 + lh*8);
        v8bf q11 = *(const v8bf*)(qp + 48 + lh*8);
        #pragma unroll
        for (int i = 0; i < 8; ++i) {
            qa0[i] = q00[i]; qa0[8+i] = q01[i];
            qa1[i] = q10[i]; qa1[8+i] = q11[i];
        }
    }

    float mrun[8], lrun[8], fac[8];
    v8f oa[4];
    #pragma unroll
    for (int r = 0; r < 8; ++r) { mrun[r] = -1e30f; lrun[r] = 0.f; }
    #pragma unroll
    for (int j = 0; j < 4; ++j) { v8f z = {}; oa[j] = z; }

    const float scale = 0.125f;

    for (int kt = 0; kt < SS/32; ++kt) {
        __syncthreads();
        // K tile: 256 x 16B chunks via async-to-LDS (2 per thread)
        #pragma unroll
        for (int i = 0; i < 2; ++i) {
            int lin = i*128 + t;
            int kr = lin >> 3, c8 = (lin & 7) * 8;
            async_ld16(&Ks[kr][c8],
                       Kb + (size_t)(b*SS + kt*32 + kr) * DM + h*DH + c8);
        }
        // V tile: transpose through VGPRs into Vt[dh][kidx]
        #pragma unroll
        for (int i = 0; i < 2; ++i) {
            int lin = i*128 + t;
            int kr = lin >> 3, c8 = (lin & 7) * 8;
            v8bf vv = *(const v8bf*)(Vb + (size_t)(b*SS + kt*32 + kr) * DM + h*DH + c8);
            #pragma unroll
            for (int j = 0; j < 8; ++j) Vt[c8 + j][kr] = vv[j];
        }
        if (t < 32) {
            int tk = kt*32 + t, g = tk % GG;
            ccK[t][0] = cc[2*(b*GG + g) + 0];
            ccK[t][1] = cc[2*(b*GG + g) + 1];
            mk[t] = maskbuf[b*SS + tk];
        }
        WAIT_ASYNC(0);
        __syncthreads();

        // scores: two 16-col blocks, dh=64 via 2 chained WMMAs
        v8f sc[2];
        #pragma unroll
        for (int cb = 0; cb < 2; ++cb) {
            const __bf16* kp = &Ks[cb*16 + lm][0];
            v16bf bb0, bb1;
            #pragma unroll
            for (int i = 0; i < 16; ++i) {
                bb0[i] = kp[     (i & 8)*2 + lh*8 + (i & 7)];
                bb1[i] = kp[32 + (i & 8)*2 + lh*8 + (i & 7)];
            }
            v8f s = {};
            s = __builtin_amdgcn_wmma_f32_16x16x32_bf16(false, qa0, false, bb0, (short)0, s, false, false);
            s = __builtin_amdgcn_wmma_f32_16x16x32_bf16(false, qa1, false, bb1, (short)0, s, false, false);
            sc[cb] = s;
        }
        // RPE bias + key mask
        #pragma unroll
        for (int cb = 0; cb < 2; ++cb) {
            int kloc = cb*16 + lm;
            float ckx = ccK[kloc][0], cky = ccK[kloc][1];
            float mv = mk[kloc];
            #pragma unroll
            for (int r = 0; r < 8; ++r) {
                int qloc = wid*16 + r + 8*lh;
                int bx = rp_bucket(ccQ[qloc][0] - ckx);
                int by = rp_bucket(ccQ[qloc][1] - cky);
                float val = sc[cb][r] * scale + rpex[bx*HH + h] + rpey[by*HH + h];
                sc[cb][r] = (mv > 0.5f) ? val : -3.0e38f;
            }
        }
        // online softmax (row's 16 cols live in one 16-lane half)
        #pragma unroll
        for (int r = 0; r < 8; ++r) {
            float mloc = fmaxf(sc[0][r], sc[1][r]);
            #pragma unroll
            for (int off = 8; off > 0; off >>= 1)
                mloc = fmaxf(mloc, __shfl_xor(mloc, off, 16));
            float mnew = fmaxf(fmaxf(mrun[r], mloc), -1e30f);
            float f  = __expf(mrun[r] - mnew);
            float p0 = __expf(sc[0][r] - mnew);
            float p1 = __expf(sc[1][r] - mnew);
            float ssum = p0 + p1;
            #pragma unroll
            for (int off = 8; off > 0; off >>= 1)
                ssum += __shfl_xor(ssum, off, 16);
            mrun[r] = mnew;
            lrun[r] = lrun[r] * f + ssum;
            fac[r] = f;
            sc[0][r] = p0; sc[1][r] = p1;
        }
        #pragma unroll
        for (int j = 0; j < 4; ++j)
            #pragma unroll
            for (int r = 0; r < 8; ++r) oa[j][r] *= fac[r];

        // C-layout P -> LDS -> A-layout fragment
        #pragma unroll
        for (int cb = 0; cb < 2; ++cb)
            #pragma unroll
            for (int r = 0; r < 8; ++r)
                Pst[wid][r + 8*lh][cb*16 + lm] = (__bf16)sc[cb][r];
        __syncthreads();

        const __bf16* pp = &Pst[wid][lm][0];
        v16bf pa;
        #pragma unroll
        for (int i = 0; i < 8; ++i) {
            pa[i]   = pp[     lh*8 + i];
            pa[8+i] = pp[16 + lh*8 + i];
        }
        #pragma unroll
        for (int j = 0; j < 4; ++j) {
            const __bf16* vp = &Vt[j*16 + lm][0];
            v16bf vb;
            #pragma unroll
            for (int i = 0; i < 16; ++i) vb[i] = vp[(i & 8)*2 + lh*8 + (i & 7)];
            oa[j] = __builtin_amdgcn_wmma_f32_16x16x32_bf16(false, pa, false, vb, (short)0, oa[j], false, false);
        }
    }

    #pragma unroll
    for (int j = 0; j < 4; ++j) {
        int col = h*DH + j*16 + lm;
        #pragma unroll
        for (int r = 0; r < 8; ++r) {
            int row = b*SS + q0 + wid*16 + r + 8*lh;
            O[(size_t)row * DM + col] = (__bf16)(oa[j][r] / lrun[r]);
        }
    }
}

// ---------------------------------------------------------------------------
__global__ __launch_bounds__(256)
void pool_kernel(const float* __restrict__ x, const float* __restrict__ ntf,
                 float* __restrict__ out)
{
    int r = blockIdx.x, d = threadIdx.x;
    int b = r / GG, g = r % GG;
    float xc = x[((size_t)b*SS +        g) * DM + d];
    float xv = x[((size_t)b*SS +  GG  + g) * DM + d];
    float xt = x[((size_t)b*SS + 2*GG + g) * DM + d];
    float p = (ntf[r] > 0.5f) ? (xc + xv) * 0.5f : (xc + xv + xt) * (1.f / 3.f);
    out[(size_t)r * DM + d] = p;
}

// ---------------------------------------------------------------------------
extern "C" void kernel_launch(void* const* d_in, const int* in_sizes, int n_in,
                              void* d_out, int out_size, void* d_ws, size_t ws_size,
                              hipStream_t stream)
{
    (void)in_sizes; (void)n_in; (void)out_size; (void)ws_size;
    const float* coords    = (const float*)d_in[0];
    const int*   class_idx = (const int*)  d_in[1];
    const float* visions   = (const float*)d_in[2];
    const float* texts     = (const float*)d_in[3];
    const unsigned char* pm= (const unsigned char*)d_in[4];
    const float* class_emb = (const float*)d_in[5];
    const float* Wv  = (const float*)d_in[6];
    const float* bv  = (const float*)d_in[7];
    const float* Wt_ = (const float*)d_in[8];
    const float* bt  = (const float*)d_in[9];
    const float* ln1s= (const float*)d_in[10];
    const float* ln1b= (const float*)d_in[11];
    const float* Wq  = (const float*)d_in[12];
    const float* bq  = (const float*)d_in[13];
    const float* Wk  = (const float*)d_in[14];
    const float* bk  = (const float*)d_in[15];
    const float* Wvp = (const float*)d_in[16];
    const float* bvp = (const float*)d_in[17];
    const float* Wo  = (const float*)d_in[18];
    const float* bo  = (const float*)d_in[19];
    const float* rpex= (const float*)d_in[20];
    const float* rpey= (const float*)d_in[21];
    const float* ln2s= (const float*)d_in[22];
    const float* ln2b= (const float*)d_in[23];
    const float* W1  = (const float*)d_in[24];
    const float* b1  = (const float*)d_in[25];
    const float* W2  = (const float*)d_in[26];
    const float* b2  = (const float*)d_in[27];

    char* p = (char*)d_ws;
    auto alloc = [&](size_t bytes) -> void* {
        void* r = (void*)p;
        p += (bytes + 255) & ~(size_t)255;
        return r;
    };
    float*  x    = (float*) alloc((size_t)NTOK * DM * 4);
    float*  cc   = (float*) alloc((size_t)NROW * 2 * 4);
    float*  tmask= (float*) alloc((size_t)NROW * 4);
    float*  ntf  = (float*) alloc((size_t)NROW * 4);
    float*  mbuf = (float*) alloc((size_t)NTOK * 4);
    __bf16* hbb  = (__bf16*)alloc((size_t)NTOK * DM * 2);
    __bf16* qbb  = (__bf16*)alloc((size_t)NTOK * DM * 2);
    __bf16* kbb  = (__bf16*)alloc((size_t)NTOK * DM * 2);
    __bf16* vbb  = (__bf16*)alloc((size_t)NTOK * DM * 2);
    __bf16* obb  = (__bf16*)alloc((size_t)NTOK * DM * 2);
    __bf16* ffb  = (__bf16*)alloc((size_t)NTOK * DFF_ * 2);
    __bf16* visb = (__bf16*)alloc((size_t)NROW * VD_ * 2);
    __bf16* texb = (__bf16*)alloc((size_t)NROW * TD_ * 2);
    __bf16* WvT  = (__bf16*)alloc((size_t)DM * VD_ * 2);
    __bf16* WtT  = (__bf16*)alloc((size_t)DM * TD_ * 2);
    __bf16* WqT  = (__bf16*)alloc((size_t)NL_ * DM * DM * 2);
    __bf16* WkT  = (__bf16*)alloc((size_t)NL_ * DM * DM * 2);
    __bf16* WvpT = (__bf16*)alloc((size_t)NL_ * DM * DM * 2);
    __bf16* WoT  = (__bf16*)alloc((size_t)NL_ * DM * DM * 2);
    __bf16* W1T  = (__bf16*)alloc((size_t)NL_ * DFF_ * DM * 2);
    __bf16* W2T  = (__bf16*)alloc((size_t)NL_ * DM * DFF_ * 2);

    // ---- one-time conversions / weight transposes (bf16, N-major) ----
    {
        int n;
        n = NROW * VD_;  cvt_bf16_kernel<<<(n+255)/256, 256, 0, stream>>>(visions, visb, n);
        n = NROW * TD_;  cvt_bf16_kernel<<<(n+255)/256, 256, 0, stream>>>(texts,   texb, n);
        n = VD_ * DM;    transpose_bf16_kernel<<<(n+255)/256, 256, 0, stream>>>(Wv,  WvT, VD_, DM);
        n = TD_ * DM;    transpose_bf16_kernel<<<(n+255)/256, 256, 0, stream>>>(Wt_, WtT, TD_, DM);
        for (int l = 0; l < NL_; ++l) {
            n = DM * DM;
            transpose_bf16_kernel<<<(n+255)/256, 256, 0, stream>>>(Wq  + (size_t)l*n, WqT  + (size_t)l*n, DM, DM);
            transpose_bf16_kernel<<<(n+255)/256, 256, 0, stream>>>(Wk  + (size_t)l*n, WkT  + (size_t)l*n, DM, DM);
            transpose_bf16_kernel<<<(n+255)/256, 256, 0, stream>>>(Wvp + (size_t)l*n, WvpT + (size_t)l*n, DM, DM);
            transpose_bf16_kernel<<<(n+255)/256, 256, 0, stream>>>(Wo  + (size_t)l*n, WoT  + (size_t)l*n, DM, DM);
            n = DM * DFF_;
            transpose_bf16_kernel<<<(n+255)/256, 256, 0, stream>>>(W1 + (size_t)l*n, W1T + (size_t)l*n, DM, DFF_);
            transpose_bf16_kernel<<<(n+255)/256, 256, 0, stream>>>(W2 + (size_t)l*n, W2T + (size_t)l*n, DFF_, DM);
        }
    }

    prep_kernel  <<<NROW, 256, 0, stream>>>(coords, texts, pm, cc, tmask, ntf);
    mask_kernel  <<<NTOK/256, 256, 0, stream>>>(pm, tmask, mbuf);
    gather_kernel<<<NROW, 256, 0, stream>>>(class_idx, class_emb, x);

    // E_vision -> x tokens [G,2G);  E_text (masked) -> x tokens [2G,3G)
    gemm_bf16t<<<dim3(NROW/64, DM/64), 256, 0, stream>>>(
        visb, WvT, bv, x, nullptr, NROW, VD_, DM, GG, SS, GG, nullptr, 0, 0);
    gemm_bf16t<<<dim3(NROW/64, DM/64), 256, 0, stream>>>(
        texb, WtT, bt, x, nullptr, NROW, TD_, DM, GG, SS, 2*GG, tmask, 0, 0);

    for (int l = 0; l < NL_; ++l) {
        layernorm_kernel<<<NTOK, 256, 0, stream>>>(x, ln1s + l*DM, ln1b + l*DM, hbb);
        gemm_bf16t<<<dim3(NTOK/64, DM/64), 256, 0, stream>>>(
            hbb, WqT + (size_t)l*DM*DM, bq + l*DM, nullptr, qbb, NTOK, DM, DM, NTOK, NTOK, 0, nullptr, 0, 0);
        gemm_bf16t<<<dim3(NTOK/64, DM/64), 256, 0, stream>>>(
            hbb, WkT + (size_t)l*DM*DM, bk + l*DM, nullptr, kbb, NTOK, DM, DM, NTOK, NTOK, 0, nullptr, 0, 0);
        gemm_bf16t<<<dim3(NTOK/64, DM/64), 256, 0, stream>>>(
            hbb, WvpT + (size_t)l*DM*DM, bvp + l*DM, nullptr, vbb, NTOK, DM, DM, NTOK, NTOK, 0, nullptr, 0, 0);

        attn_kernel<<<dim3(SS/64, HH, BSZ), 128, 0, stream>>>(
            qbb, kbb, vbb, obb, cc, mbuf,
            rpex + (size_t)l*(2*NB_-1)*HH, rpey + (size_t)l*(2*NB_-1)*HH);

        gemm_bf16t<<<dim3(NTOK/64, DM/64), 256, 0, stream>>>(
            obb, WoT + (size_t)l*DM*DM, bo + l*DM, x, nullptr, NTOK, DM, DM, NTOK, NTOK, 0, nullptr, 1, 0);

        layernorm_kernel<<<NTOK, 256, 0, stream>>>(x, ln2s + l*DM, ln2b + l*DM, hbb);
        gemm_bf16t<<<dim3(NTOK/64, DFF_/64), 256, 0, stream>>>(
            hbb, W1T + (size_t)l*DFF_*DM, b1 + l*DFF_, nullptr, ffb, NTOK, DM, DFF_, NTOK, NTOK, 0, nullptr, 0, 1);
        gemm_bf16t<<<dim3(NTOK/64, DM/64), 256, 0, stream>>>(
            ffb, W2T + (size_t)l*DM*DFF_, b2 + l*DM, x, nullptr, NTOK, DFF_, DM, NTOK, NTOK, 0, nullptr, 1, 0);
    }

    pool_kernel<<<NROW, 256, 0, stream>>>(x, ntf, (float*)d_out);
}
